// Attentions_26946624815570
// MI455X (gfx1250) — matrix operand order repaired and checked by reading, not compiled, not run on hardware
//
#include <hip/hip_runtime.h>
#include <hip/hip_bf16.h>

typedef _Float16 f16_t;
typedef __attribute__((ext_vector_type(16))) _Float16 v16h;
typedef __attribute__((ext_vector_type(8)))  _Float16 v8h;
typedef __attribute__((ext_vector_type(8)))  float    v8f;
typedef __attribute__((ext_vector_type(4)))  float    v4f;
typedef __attribute__((ext_vector_type(4)))  unsigned int u32x4;
typedef __attribute__((ext_vector_type(8)))  int          i32x8;
typedef __attribute__((ext_vector_type(4)))  int          i32x4;

#define N_HEAD 8
#define HID    64
#define IN_DIM 512
#define OUT_DIM 512
#define BATCH  2
#define LSEQ   2048
#define ROWS   (BATCH * LSEQ)   // 4096

#if __has_builtin(__builtin_amdgcn_tensor_load_to_lds) && __has_builtin(__builtin_amdgcn_s_wait_tensorcnt)
#define USE_TDM 1
#else
#define USE_TDM 0
#endif

// ---------------- workspace layout (bytes) ----------------
#define OFF_WVT   0u                                  // Wv^T  512x512 f16 (NxK)
#define OFF_WDT   (OFF_WVT  + 512u*512u*2u)           // Wd^T  512x512 f16 (NxK)
#define OFF_WQE   (OFF_WDT  + 512u*512u*2u)           // 512*8 f32
#define OFF_WKE   (OFF_WQE  + 512u*8u*4u)             // 512*8 f32
#define OFF_BQE   (OFF_WKE  + 512u*8u*4u)             // 8 f32 (pad 256)
#define OFF_BKE   (OFF_BQE  + 256u)                   // 8 f32 (pad 256)
#define OFF_QW    (OFF_BKE  + 256u)                   // 16*2048 f32
#define OFF_KW    (OFF_QW   + 16u*2048u*4u)           // 16*2048 f32
#define OFF_VXT   (OFF_KW   + 16u*2048u*4u)           // vx^T: (hb*64+d)*2048+l  f16
#define OFF_OUTH  (OFF_VXT  + 16u*64u*2048u*2u)       // 4096*512 f16 row-major

// ---------------- WMMA helpers ----------------
__device__ __forceinline__ v8f wmma16x16x32(v16h a, v16h b, v8f c) {
  return __builtin_amdgcn_wmma_f32_16x16x32_f16(false, a, false, b,
                                                (short)0, c, false, false);
}

__device__ __forceinline__ v16h combine16(v8h lo, v8h hi) {
  v16h r;
#pragma unroll
  for (int j = 0; j < 8; ++j) { r[j] = lo[j]; r[8 + j] = hi[j]; }
  return r;
}

// A fragment 16x32 from row-major f32 (cvt to f16); fully vectorized (4x b128)
__device__ __forceinline__ v16h load_a_f32(const float* A, int lda,
                                           int row0, int k0, int lane) {
  const int row = row0 + (lane & 15);
  const int kh  = lane >> 4;
  const float* p = A + (size_t)row * lda + k0 + kh * 8;
  v4f x0 = *(const v4f*)(p);
  v4f x1 = *(const v4f*)(p + 4);
  v4f x2 = *(const v4f*)(p + 16);
  v4f x3 = *(const v4f*)(p + 20);
  v16h a;
#pragma unroll
  for (int j = 0; j < 4; ++j) {
    a[j]      = (f16_t)x0[j];
    a[4 + j]  = (f16_t)x1[j];
    a[8 + j]  = (f16_t)x2[j];
    a[12 + j] = (f16_t)x3[j];
  }
  return a;
}

// A fragment 16x32 from row-major f16 (2x b128)
__device__ __forceinline__ v16h load_a_f16(const f16_t* A, int lda,
                                           int row0, int k0, int lane) {
  const int row = row0 + (lane & 15);
  const int kh  = lane >> 4;
  const f16_t* p = A + (size_t)row * lda + k0 + kh * 8;
  return combine16(*(const v8h*)p, *(const v8h*)(p + 16));
}

// B fragment 32x16 from TRANSPOSED (NxK, K-contiguous) f16 (2x b128)
__device__ __forceinline__ v16h load_bT_f16(const f16_t* BT, int ldk,
                                            int k0, int col0, int lane) {
  const int col = col0 + (lane & 15);
  const int kh  = lane >> 4;
  const f16_t* p = BT + (size_t)col * ldk + k0 + kh * 8;
  return combine16(*(const v8h*)p, *(const v8h*)(p + 16));
}

__device__ __forceinline__ float fast_tanh(float x) {
  float e2 = __expf(2.f * x);
  return 1.f - 2.f * __builtin_amdgcn_rcpf(e2 + 1.f);  // saturates to +-1
}

// ---------------- kernel 1: fold w_score into projection weights ----------------
__global__ void prep_eff_kernel(const float* __restrict__ W,
                                const float* __restrict__ bvec,
                                const float* __restrict__ wsc,
                                float* __restrict__ w_eff,
                                float* __restrict__ b_eff) {
  int idx = blockIdx.x * blockDim.x + threadIdx.x;   // 0..4095
  if (idx < 512 * 8) {
    int i = idx >> 3, h = idx & 7;
    float s = 0.f;
    const float* wp = W + (size_t)i * 512 + h * 64;
#pragma unroll 8
    for (int d = 0; d < 64; ++d) s += wp[d] * wsc[d];
    w_eff[i * 8 + h] = s;
  }
  if (idx < 8) {
    float s = 0.f;
    const float* bp = bvec + idx * 64;
#pragma unroll 8
    for (int d = 0; d < 64; ++d) s += bp[d] * wsc[d];
    b_eff[idx] = s;
  }
}

// ---------------- kernel 2: tiled transpose + f32->f16 (W -> W^T) ----------------
// dst[n*512+k] = (f16) src[k*512+n];  32x32 tiles via LDS, coalesced both sides
__global__ void transpose_cvt_kernel(const float* __restrict__ src,
                                     f16_t* __restrict__ dst) {
  __shared__ float tile[32][33];
  const int bk = blockIdx.x * 32;   // k-range of source rows
  const int bn = blockIdx.y * 32;   // n-range of source cols
  const int tx = threadIdx.x & 31;
  const int ty = threadIdx.x >> 5;  // 0..7
#pragma unroll
  for (int i = 0; i < 32; i += 8)
    tile[ty + i][tx] = src[(size_t)(bk + ty + i) * 512 + bn + tx];
  __syncthreads();
#pragma unroll
  for (int i = 0; i < 32; i += 8)
    dst[(size_t)(bn + ty + i) * 512 + bk + tx] = (f16_t)tile[tx][ty + i];
}

// ---------------- kernel 3: qw / kw skinny GEMMs ----------------
__global__ void qwkw_kernel(const float* __restrict__ q, const float* __restrict__ k,
                            const float* __restrict__ wqe, const float* __restrict__ wke,
                            const float* __restrict__ bqe, const float* __restrict__ bke,
                            float* __restrict__ qw, float* __restrict__ kw) {
  int row = blockIdx.x * blockDim.x + threadIdx.x;   // 0..4095 (b*2048+l)
  float aq[8], ak[8];
#pragma unroll
  for (int h = 0; h < 8; ++h) { aq[h] = 0.f; ak[h] = 0.f; }
  const float* qr = q + (size_t)row * IN_DIM;
  const float* kr = k + (size_t)row * IN_DIM;
#pragma unroll 4
  for (int i = 0; i < IN_DIM; ++i) {
    float qv = qr[i], kv = kr[i];
    const float* w8q = wqe + i * 8;
    const float* w8k = wke + i * 8;
#pragma unroll
    for (int h = 0; h < 8; ++h) { aq[h] += qv * w8q[h]; ak[h] += kv * w8k[h]; }
  }
  int b = row >> 11, l = row & 2047;
#pragma unroll
  for (int h = 0; h < 8; ++h) {
    qw[((h * 2 + b) << 11) + l] = aq[h] + bqe[h];
    kw[((h * 2 + b) << 11) + l] = ak[h] + bke[h];
  }
}

// ---------------- kernel 4: V projection GEMM (WMMA), writes vx^T ----------------
// vxT[(hb*64+d)*2048 + l] = (f16)( (v @ Wv)[b*2048+l, h*64+d] + bv )
__global__ void vproj_kernel(const float* __restrict__ V,
                             const f16_t* __restrict__ WvT,
                             const float* __restrict__ bv,
                             f16_t* __restrict__ vxT) {
  const int lane = threadIdx.x & 31;
  const int wave = threadIdx.x >> 5;
  const int row0 = blockIdx.x * 128 + wave * 16;
  const int col0 = blockIdx.y * 64;
  v8f acc[4];
  const v8f zero = {};
#pragma unroll
  for (int t = 0; t < 4; ++t) acc[t] = zero;

  for (int k0 = 0; k0 < IN_DIM; k0 += 32) {
    v16h a = load_a_f32(V, IN_DIM, row0, k0, lane);
#pragma unroll
    for (int t = 0; t < 4; ++t) {
      v16h b = load_bT_f16(WvT, IN_DIM, k0, col0 + t * 16, lane);
      acc[t] = wmma16x16x32(a, b, acc[t]);
    }
  }
  const int mh = lane >> 4;
  const int l0g = row0 + mh * 8;          // 8 consecutive rows (same batch b)
  const int bb = l0g >> 11, l0 = l0g & 2047;
#pragma unroll
  for (int t = 0; t < 4; ++t) {
    int col = col0 + t * 16 + (lane & 15);
    int h = col >> 6, d = col & 63;
    float bvv = bv[col];
    v8h pk;
#pragma unroll
    for (int r = 0; r < 8; ++r) pk[r] = (f16_t)(acc[t][r] + bvv);
    *(v8h*)(vxT + ((size_t)((h * 2 + bb) * 64 + d)) * 2048 + l0) = pk;
  }
}

// ---------------- kernel 5: fused score + softmax + P@V (WMMA) ----------------
// block = 256 thr (8 waves), covers 128 q rows of one (h,b); grid (16, 16)
__global__ void attn_kernel(const float* __restrict__ qw, const float* __restrict__ kw,
                            const f16_t* __restrict__ vxT,
                            float* __restrict__ score, f16_t* __restrict__ outh) {
  __shared__ float s_kw[LSEQ];
  __shared__ float s_den[128];
  __shared__ f16_t s_tile[8 * 16 * 64];   // per-wave 16x64 f16 transpose staging
  const int hb = blockIdx.y;              // h*2+b (matches score_flat order)
  const int h = hb >> 1, b = hb & 1;
  const int lane = threadIdx.x & 31;
  const int wave = threadIdx.x >> 5;
  const float* kwp = kw + (size_t)hb * LSEQ;
  const float* qwp = qw + (size_t)hb * LSEQ;

#if USE_TDM
  // Tensor Data Mover: DMA the 8KB kw row into LDS (wave 0 issues once).
  if (wave == 0) {
    unsigned int lds_off = (unsigned int)(uintptr_t)&s_kw[0];  // low 32b = LDS offset
    unsigned long long ga = (unsigned long long)(uintptr_t)kwp;
    u32x4 g0;
    g0[0] = 1u;                                         // count=1 (valid user D#)
    g0[1] = lds_off;                                    // lds_addr
    g0[2] = (unsigned int)(ga & 0xFFFFFFFFu);           // global_addr[31:0]
    g0[3] = (unsigned int)((ga >> 32) & 0x1FFFFFFu) | (2u << 30); // [56:32] | type=2
    i32x8 g1;
    g1[0] = (2 << 16);                 // workgroup_mask=0, data_size=2 (4B)
    g1[1] = (int)(2048u << 16);        // tensor_dim0[15:0] in bits[31:16]
    g1[2] = (1 << 16);                 // tensor_dim0 hi=0 | tensor_dim1=1
    g1[3] = (int)(2048u << 16);        // tile_dim0 = 2048
    g1[4] = 1;                         // tile_dim1 = 1, tile_dim2 = 0
    g1[5] = 2048;                      // tensor_dim0_stride low32
    g1[6] = (int)(2048u << 16);        // stride0 hi=0 | tensor_dim1_stride lo16
    g1[7] = 0;
    i32x4 gz4 = {};
    i32x8 gz8 = {};
    // 6-arg toolchain form: (g0, g1, g2, g3, g_extra, cpol)
    __builtin_amdgcn_tensor_load_to_lds(g0, g1, gz4, gz4, gz8, 0);
    __builtin_amdgcn_s_wait_tensorcnt(0);
  }
  __syncthreads();
#else
  for (int i = threadIdx.x; i < LSEQ; i += 256) s_kw[i] = kwp[i];
  __syncthreads();
#endif

  const int qbase = blockIdx.x * 128 + wave * 16;

  // ---- pass 1: denominators (tanh in (-1,1) => no max subtraction needed) ----
  float qwr[16], ps[16];
#pragma unroll
  for (int r = 0; r < 16; ++r) { qwr[r] = qwp[qbase + r]; ps[r] = 0.f; }
  for (int k = lane; k < LSEQ; k += 32) {
    float kv = s_kw[k];
#pragma unroll
    for (int r = 0; r < 16; ++r) ps[r] += __expf(fast_tanh(qwr[r] + kv));
  }
#pragma unroll
  for (int r = 0; r < 16; ++r) {
    float s = ps[r];
#pragma unroll
    for (int off = 16; off > 0; off >>= 1) s += __shfl_xor(s, off, 32);
    if (lane == 0) s_den[wave * 16 + r] = s;
  }
  __syncthreads();

  // ---- pass 2: normalized probs -> score_flat (vector stores) + WMMA P@V ----
  const int row = lane & 15;
  const int kh  = lane >> 4;
  const float invd = 1.f / s_den[wave * 16 + row];
  const float qv = qwp[qbase + row];
  float* srow = score + ((size_t)(hb * LSEQ + qbase + row)) * LSEQ;
  const f16_t* vbT = vxT + (size_t)hb * HID * LSEQ;

  v8f acc[4];
  const v8f zero = {};
#pragma unroll
  for (int t = 0; t < 4; ++t) acc[t] = zero;

  for (int kb = 0; kb < LSEQ; kb += 32) {
    float f0[8], f1[8];
#pragma unroll
    for (int j = 0; j < 8; ++j) {
      f0[j] = __expf(fast_tanh(qv + s_kw[kb + kh * 8 + j])) * invd;
      f1[j] = __expf(fast_tanh(qv + s_kw[kb + 16 + kh * 8 + j])) * invd;
    }
    v16h a;
    v4f s0a, s0b, s1a, s1b;
#pragma unroll
    for (int j = 0; j < 4; ++j) {
      a[j]      = (f16_t)f0[j];     s0a[j] = f0[j];
      a[4 + j]  = (f16_t)f0[4 + j]; s0b[j] = f0[4 + j];
      a[8 + j]  = (f16_t)f1[j];     s1a[j] = f1[j];
      a[12 + j] = (f16_t)f1[4 + j]; s1b[j] = f1[4 + j];
    }
    *(v4f*)(srow + kb + kh * 8)          = s0a;
    *(v4f*)(srow + kb + kh * 8 + 4)      = s0b;
    *(v4f*)(srow + kb + 16 + kh * 8)     = s1a;
    *(v4f*)(srow + kb + 16 + kh * 8 + 4) = s1b;

    if (kb + 32 < LSEQ)
      __builtin_prefetch(vbT + kb + 32, 0, 0);
#pragma unroll
    for (int t = 0; t < 4; ++t) {
      v16h bf = load_bT_f16(vbT, LSEQ, kb, t * 16, lane);
      acc[t] = wmma16x16x32(a, bf, acc[t]);
    }
  }

  // ---- epilogue: per-wave LDS transpose -> 64B-contiguous global stores ----
  {
    f16_t* tp = s_tile + wave * 1024;
#pragma unroll
    for (int t = 0; t < 4; ++t)
#pragma unroll
      for (int r = 0; r < 8; ++r)
        tp[(kh * 8 + r) * 64 + t * 16 + (lane & 15)] = (f16_t)acc[t][r];
    __syncthreads();
    const int rrow = lane & 15;
    const int half = lane >> 4;
    const f16_t* rp = s_tile + wave * 1024 + rrow * 64 + half * 32;
    f16_t* op = outh + ((size_t)(b * LSEQ + qbase + rrow)) * 512 + h * 64 + half * 32;
#pragma unroll
    for (int c = 0; c < 32; c += 8)
      *(v8h*)(op + c) = *(const v8h*)(rp + c);
  }
}

// ---------------- kernel 6: output GEMM, flipped: out^T = Wd^T @ outh^T ----------
// A = WdT (NxK, K-contig), B = outh (row-major => K-contig per out-row),
// D stores are 8-consecutive-f32 runs per lane.
__global__ void outgemm_kernel(const f16_t* __restrict__ outh,
                               const f16_t* __restrict__ WdT,
                               const float* __restrict__ bd,
                               float* __restrict__ out) {
  const int lane = threadIdx.x & 31;
  const int wave = threadIdx.x >> 5;
  const int m0 = blockIdx.x * 128 + wave * 16;  // out-column tile (of 512)
  const int n0 = blockIdx.y * 64;               // out-row strip (of 4096)
  v8f acc[4];
  const v8f zero = {};
#pragma unroll
  for (int t = 0; t < 4; ++t) acc[t] = zero;

  for (int k0 = 0; k0 < 512; k0 += 32) {
    v16h a = load_a_f16(WdT, 512, m0, k0, lane);   // A(m,k) = Wd^T
#pragma unroll
    for (int t = 0; t < 4; ++t) {
      // B(k,n) = outh[n,k]: per-lane fixed n, contiguous k
      v16h b = load_bT_f16(outh, 512, k0, n0 + t * 16, lane);
      acc[t] = wmma16x16x32(a, b, acc[t]);
    }
  }
  const int mh = lane >> 4;
  const int mbase = m0 + mh * 8;
  v4f b0 = *(const v4f*)(bd + mbase);
  v4f b1 = *(const v4f*)(bd + mbase + 4);
#pragma unroll
  for (int t = 0; t < 4; ++t) {
    int n = n0 + t * 16 + (lane & 15);            // out row
    v4f lo, hi;
#pragma unroll
    for (int r = 0; r < 4; ++r) { lo[r] = acc[t][r]; hi[r] = acc[t][4 + r]; }
    lo += b0; hi += b1;
    float* op = out + (size_t)n * OUT_DIM + mbase;
    *(v4f*)op = lo;
    *(v4f*)(op + 4) = hi;
  }
}

// ---------------- launch ----------------
extern "C" void kernel_launch(void* const* d_in, const int* in_sizes, int n_in,
                              void* d_out, int out_size, void* d_ws, size_t ws_size,
                              hipStream_t stream) {
  const float* q   = (const float*)d_in[0];
  const float* k   = (const float*)d_in[1];
  const float* v   = (const float*)d_in[2];
  const float* Wq  = (const float*)d_in[3];
  const float* bq  = (const float*)d_in[4];
  const float* Wk  = (const float*)d_in[5];
  const float* bk  = (const float*)d_in[6];
  const float* Wv  = (const float*)d_in[7];
  const float* bv  = (const float*)d_in[8];
  const float* wsc = (const float*)d_in[9];
  const float* Wd  = (const float*)d_in[10];
  const float* bd  = (const float*)d_in[11];

  float* out   = (float*)d_out;
  float* score = out + (size_t)BATCH * LSEQ * OUT_DIM;

  char* ws = (char*)d_ws;
  f16_t* WvT  = (f16_t*)(ws + OFF_WVT);
  f16_t* WdT  = (f16_t*)(ws + OFF_WDT);
  float* wqe  = (float*)(ws + OFF_WQE);
  float* wke  = (float*)(ws + OFF_WKE);
  float* bqe  = (float*)(ws + OFF_BQE);
  float* bke  = (float*)(ws + OFF_BKE);
  float* qwg  = (float*)(ws + OFF_QW);
  float* kwg  = (float*)(ws + OFF_KW);
  f16_t* vxT  = (f16_t*)(ws + OFF_VXT);
  f16_t* outh = (f16_t*)(ws + OFF_OUTH);

  prep_eff_kernel<<<16, 256, 0, stream>>>(Wq, bq, wsc,      wqe, bqe);
  prep_eff_kernel<<<16, 256, 0, stream>>>(Wk, bk, wsc + 64, wke, bke);
  transpose_cvt_kernel<<<dim3(16, 16), 256, 0, stream>>>(Wv, WvT);
  transpose_cvt_kernel<<<dim3(16, 16), 256, 0, stream>>>(Wd, WdT);
  qwkw_kernel<<<ROWS / 256, 256, 0, stream>>>(q, k, wqe, wke, bqe, bke, qwg, kwg);
  vproj_kernel<<<dim3(ROWS / 128, IN_DIM / 64), 256, 0, stream>>>(v, WvT, bv, vxT);
  attn_kernel<<<dim3(LSEQ / 128, N_HEAD * BATCH), 256, 0, stream>>>(qwg, kwg, vxT, score, outh);
  outgemm_kernel<<<dim3(OUT_DIM / 128, ROWS / 64), 256, 0, stream>>>(outh, WdT, bd, out);
}
